// NestedGIN_8632884265156
// MI455X (gfx1250) — matrix operand order repaired and checked by reading, not compiled, not run on hardware
//
#include <hip/hip_runtime.h>
#include <hip/hip_bf16.h>

// ---------------------------------------------------------------------------
// NestedGIN forward for MI455X (gfx1250, wave32, WMMA).
// GEMMs run on v_wmma_f32_16x16x32_bf16. Activations feeding a GEMM are
// written as bf16 by the preceding elementwise pass (halves A traffic and
// removes all cvt VALU from the WMMA loop). Weights are pre-packed once per
// call into the exact B-fragment lane layout (ISA 7.12.2) so each B fragment
// is one contiguous 32B load.
// ---------------------------------------------------------------------------

typedef __attribute__((ext_vector_type(16))) __bf16 v16bf;
typedef __attribute__((ext_vector_type(8)))  __bf16 v8bf;
typedef __attribute__((ext_vector_type(8)))  float  v8f;

#define HID 128
#define NCLS 10

__device__ __forceinline__ unsigned short f2bf_bits(float f) {
    unsigned u = __builtin_bit_cast(unsigned, f);
    unsigned r = (u + 0x7FFFu + ((u >> 16) & 1u)) >> 16;   // round-to-nearest-even
    return (unsigned short)r;
}

// ------------------------------- utility -----------------------------------
__global__ void fillz_kernel(float* __restrict__ p, long long n) {
    long long i = (long long)blockIdx.x * blockDim.x + threadIdx.x;
    long long stride = (long long)gridDim.x * blockDim.x;
    for (; i < n; i += stride) p[i] = 0.0f;
}

// Pack fp32 weights [128(out) x 128(in)] (torch Linear layout) into bf16
// B-fragments.  Linear packed index t = ((ct*4+kc)*32 + lane)*16 + i maps to
// W[n,k] with n = ct*16 + (lane&15), k = kc*32 + 8*(lane>>4) + i (i<8), kc*32 + 16 + 8*(lane>>4) + (i-8) (i>=8).
__global__ void pack_weights_kernel(const float* __restrict__ W1,
                                    const float* __restrict__ W2,
                                    unsigned short* __restrict__ wp,
                                    int nmat_half) {
    long long idx = (long long)blockIdx.x * blockDim.x + threadIdx.x;
    long long total = (long long)nmat_half * 2 * HID * HID;
    if (idx >= total) return;
    int m = (int)(idx >> 14);                 // matrix id (16384 elems each)
    int t = (int)(idx & 16383);
    int i    =  t        & 15;
    int lane = (t >> 4)  & 31;
    int kc   = (t >> 9)  & 3;
    int ct   =  t >> 11;
    int n = ct * 16 + (lane & 15);
    int k = kc * 32 + 8 * (lane >> 4) + (i < 8 ? i : 8 + i);
    const float* Wsrc = (m < nmat_half) ? (W1 + (size_t)m * HID * HID)
                                        : (W2 + (size_t)(m - nmat_half) * HID * HID);
    wp[idx] = f2bf_bits(Wsrc[(size_t)n * HID + k]);
}

// --------------------------- graph aggregation -----------------------------
// 4 floats per thread: i in [0, E*32)
__global__ void scatter_agg_kernel(const float* __restrict__ X,
                                   const int* __restrict__ src,
                                   const int* __restrict__ dst,
                                   float* __restrict__ agg, long long total) {
    long long i = (long long)blockIdx.x * blockDim.x + threadIdx.x;
    long long stride = (long long)gridDim.x * blockDim.x;
    for (; i < total; i += stride) {
        long long e = i >> 5;
        int c = (int)(i & 31) * 4;
        const float4 v = *(const float4*)(X + (size_t)src[e] * HID + c);
        float* a = agg + (size_t)dst[e] * HID + c;
        atomicAdd(a + 0, v.x);
        atomicAdd(a + 1, v.y);
        atomicAdd(a + 2, v.z);
        atomicAdd(a + 3, v.w);
    }
}

// h0 = (1+eps)*X + agg, written as bf16 (feeds GEMM1 only)
__global__ void combine_eps_bf16_kernel(const float* __restrict__ X,
                                        const float* __restrict__ agg,
                                        const float* __restrict__ eps, int l,
                                        unsigned short* __restrict__ out,
                                        long long total) {
    float e1 = 1.0f + eps[l];
    long long i = (long long)blockIdx.x * blockDim.x + threadIdx.x;
    long long stride = (long long)gridDim.x * blockDim.x;
    for (; i < total; i += stride) out[i] = f2bf_bits(e1 * X[i] + agg[i]);
}

// ------------------------------- WMMA GEMM ---------------------------------
// Y[n, o] = sum_k X[n,k] * W[o,k] + bias[o];  X bf16 [nrows,128] row-major,
// Wp packed bf16 fragments, Y fp32 [nrows,128].
// Block = 8 wave32s = 32 rows x 128 cols; each wave: 16 rows x 32 cols.
__global__ __launch_bounds__(256) void gemm_wmma_kernel(
        const unsigned short* __restrict__ Xbf,
        const unsigned short* __restrict__ Wp,
        const float* __restrict__ bias, float* __restrict__ Y, int nrows) {
    int wave = threadIdx.x >> 5;
    int lane = threadIdx.x & 31;
    int rtile = wave >> 2;                 // 0..1
    int cgrp  = wave & 3;                  // 0..3
    int row0 = blockIdx.x * 32 + rtile * 16;
    if (row0 >= nrows) return;             // wave-uniform: EXEC stays all-ones
    int m = lane & 15, half = lane >> 4;
    const unsigned short* xrow = Xbf + (size_t)(row0 + m) * HID;
    int ct0 = cgrp * 2, ct1 = ct0 + 1;

    v8f acc0 = {}, acc1 = {};
    #pragma unroll
    for (int kc = 0; kc < 4; ++kc) {
        // A fragment (16x32 bf16): elements 0..7  <- K = kc*32 + 8*half + {0..7}
        //                          elements 8..15 <- K = kc*32 + 16 + 8*half + {0..7}
        v8bf lo = *(const v8bf*)(xrow + kc * 32 + 8 * half);
        v8bf hi = *(const v8bf*)(xrow + kc * 32 + 16 + 8 * half);
        v16bf a = __builtin_shufflevector(lo, hi, 0, 1, 2, 3, 4, 5, 6, 7,
                                                  8, 9, 10, 11, 12, 13, 14, 15);
        v16bf b0 = *(const v16bf*)(Wp + ((size_t)(ct0 * 4 + kc) * 32 + lane) * 16);
        v16bf b1 = *(const v16bf*)(Wp + ((size_t)(ct1 * 4 + kc) * 32 + lane) * 16);
        acc0 = __builtin_amdgcn_wmma_f32_16x16x32_bf16(false, a, false, b0,
                                                       (short)0, acc0, false, false);
        acc1 = __builtin_amdgcn_wmma_f32_16x16x32_bf16(false, a, false, b1,
                                                       (short)0, acc1, false, false);
    }
    // C/D layout: element r of v8f -> row M = r + 8*half, col N = lane&15
    int n = lane & 15;
    float bv0 = bias[ct0 * 16 + n];
    float bv1 = bias[ct1 * 16 + n];
    #pragma unroll
    for (int r = 0; r < 8; ++r) {
        int row = row0 + r + 8 * half;
        Y[(size_t)row * HID + ct0 * 16 + n] = acc0[r] + bv0;
        Y[(size_t)row * HID + ct1 * 16 + n] = acc1[r] + bv1;
    }
}

// ------------------------------ batch norm ---------------------------------
// Coalesced row-major partial reduction into global per-channel accumulators.
__global__ __launch_bounds__(256) void bn_partial_kernel(
        const float* __restrict__ Y, long long total,
        float* __restrict__ gsum, float* __restrict__ gsq) {
    __shared__ float ls[256], lq[256];
    int t = threadIdx.x;
    float s = 0.f, q = 0.f;
    long long i = (long long)blockIdx.x * blockDim.x + t;
    long long stride = (long long)gridDim.x * blockDim.x;   // multiple of 128
    for (; i < total; i += stride) {
        float v = Y[i];
        s += v; q += v * v;
    }
    ls[t] = s; lq[t] = q;
    __syncthreads();
    if (t < 128) {   // threads t and t+128 share channel c = t
        atomicAdd(&gsum[t], ls[t] + ls[t + 128]);
        atomicAdd(&gsq[t],  lq[t] + lq[t + 128]);
    }
}

__global__ void bn_finalize_kernel(const float* __restrict__ gsum,
                                   const float* __restrict__ gsq, int nrows,
                                   float* __restrict__ mu, float* __restrict__ rs) {
    int c = threadIdx.x;
    if (c < HID) {
        float mean = gsum[c] / (float)nrows;
        float var  = gsq[c] / (float)nrows - mean * mean;   // biased (training BN)
        mu[c] = mean;
        rs[c] = rsqrtf(var + 1e-5f);
    }
}

// BN + ReLU, bf16 output (feeds next GEMM only)
__global__ void bn_apply_relu_bf16_kernel(const float* __restrict__ Y,
                                          const float* __restrict__ mu,
                                          const float* __restrict__ rs,
                                          const float* __restrict__ g,
                                          const float* __restrict__ be,
                                          unsigned short* __restrict__ out,
                                          long long total) {
    long long i = (long long)blockIdx.x * blockDim.x + threadIdx.x;
    long long stride = (long long)gridDim.x * blockDim.x;
    for (; i < total; i += stride) {
        int c = (int)(i & 127);
        float v = (Y[i] - mu[c]) * rs[c] * g[c] + be[c];
        out[i] = f2bf_bits(fmaxf(v, 0.0f));
    }
}

// BN + ReLU, f32 output (layer output: feeds scatter + segment pooling)
__global__ void bn_apply_relu_f32_kernel(const float* __restrict__ Y,
                                         const float* __restrict__ mu,
                                         const float* __restrict__ rs,
                                         const float* __restrict__ g,
                                         const float* __restrict__ be,
                                         float* __restrict__ out, long long total) {
    long long i = (long long)blockIdx.x * blockDim.x + threadIdx.x;
    long long stride = (long long)gridDim.x * blockDim.x;
    for (; i < total; i += stride) {
        int c = (int)(i & 127);
        float v = (Y[i] - mu[c]) * rs[c] * g[c] + be[c];
        out[i] = fmaxf(v, 0.0f);
    }
}

// ----------------------------- segment means -------------------------------
__global__ void count_seg_kernel(const int* __restrict__ seg, int n,
                                 float* __restrict__ cnt) {
    long long i = (long long)blockIdx.x * blockDim.x + threadIdx.x;
    long long stride = (long long)gridDim.x * blockDim.x;
    for (; i < n; i += stride) atomicAdd(&cnt[seg[i]], 1.0f);
}

// 4 floats per thread: i in [0, N*32)
__global__ void seg_accum_kernel(const float* __restrict__ H,
                                 const int* __restrict__ n2s,
                                 float* __restrict__ subsum, int l, long long total) {
    long long i = (long long)blockIdx.x * blockDim.x + threadIdx.x;
    long long stride = (long long)gridDim.x * blockDim.x;
    for (; i < total; i += stride) {
        long long n = i >> 5;
        int c = (int)(i & 31) * 4;
        const float4 v = *(const float4*)(H + (size_t)n * HID + c);
        float* s = subsum + (size_t)n2s[n] * (4 * HID) + l * HID + c;
        atomicAdd(s + 0, v.x);
        atomicAdd(s + 1, v.y);
        atomicAdd(s + 2, v.z);
        atomicAdd(s + 3, v.w);
    }
}

__global__ void seg_div_kernel(float* __restrict__ sums,
                               const float* __restrict__ cnt, long long total) {
    long long i = (long long)blockIdx.x * blockDim.x + threadIdx.x;
    long long stride = (long long)gridDim.x * blockDim.x;
    for (; i < total; i += stride)
        sums[i] /= fmaxf(cnt[i >> 9], 1.0f);     // 512 features per segment
}

__global__ void graph_accum_kernel(const float* __restrict__ submean,
                                   const int* __restrict__ s2g,
                                   float* __restrict__ gsum, long long total) {
    long long i = (long long)blockIdx.x * blockDim.x + threadIdx.x;
    long long stride = (long long)gridDim.x * blockDim.x;
    for (; i < total; i += stride) {
        long long s = i >> 9;
        int j = (int)(i & 511);
        atomicAdd(&gsum[(size_t)s2g[s] * 512 + j], submean[i]);
    }
}

// ------------------------------- head / MLP --------------------------------
__global__ __launch_bounds__(128) void head_kernel(
        const float* __restrict__ gmean,     // [G, 512]
        const float* __restrict__ W1h,       // [128, 512]
        const float* __restrict__ b1h,       // [128]
        const float* __restrict__ W2h,       // [10, 128]
        const float* __restrict__ b2h,       // [10]
        float* __restrict__ out) {           // [G, 10] log-softmax
    int g = blockIdx.x;
    int t = threadIdx.x;
    __shared__ float h1[128];
    __shared__ float z[NCLS];
    __shared__ float lse;
    const float* gv = gmean + (size_t)g * 512;
    float acc = b1h[t];
    for (int k = 0; k < 512; ++k) acc += gv[k] * W1h[(size_t)t * 512 + k];
    h1[t] = fmaxf(acc, 0.0f);
    __syncthreads();
    if (t < NCLS) {
        float a = b2h[t];
        for (int k = 0; k < 128; ++k) a += h1[k] * W2h[(size_t)t * 128 + k];
        z[t] = a;
    }
    __syncthreads();
    if (t == 0) {
        float mx = z[0];
        for (int c = 1; c < NCLS; ++c) mx = fmaxf(mx, z[c]);
        float s = 0.f;
        for (int c = 0; c < NCLS; ++c) s += expf(z[c] - mx);
        lse = mx + logf(s);
    }
    __syncthreads();
    if (t < NCLS) out[(size_t)g * NCLS + t] = z[t] - lse;
}

// ------------------------------- launcher ----------------------------------
static inline unsigned blocks_for(long long total, int bs) {
    return (unsigned)((total + bs - 1) / bs);
}

extern "C" void kernel_launch(void* const* d_in, const int* in_sizes, int n_in,
                              void* d_out, int out_size, void* d_ws, size_t ws_size,
                              hipStream_t stream) {
    (void)n_in; (void)ws_size;
    const float* x       = (const float*)d_in[0];
    const int*   ei      = (const int*)  d_in[1];
    const int*   n2s     = (const int*)  d_in[2];
    const int*   s2g     = (const int*)  d_in[3];
    const float* W1      = (const float*)d_in[4];
    const float* b1      = (const float*)d_in[5];
    const float* g1      = (const float*)d_in[6];
    const float* be1     = (const float*)d_in[7];
    const float* W2      = (const float*)d_in[8];
    const float* b2      = (const float*)d_in[9];
    const float* g2      = (const float*)d_in[10];
    const float* be2     = (const float*)d_in[11];
    const float* eps     = (const float*)d_in[12];
    const float* lin1_W  = (const float*)d_in[13];
    const float* lin1_b  = (const float*)d_in[14];
    const float* lin2_W  = (const float*)d_in[15];
    const float* lin2_b  = (const float*)d_in[16];
    float* out = (float*)d_out;

    const int N = in_sizes[0] / HID;
    const int E = in_sizes[1] / 2;
    const int S = in_sizes[3];
    const int L = in_sizes[12];
    const int G = out_size / NCLS;
    const long long NH = (long long)N * HID;

    // ---------------- workspace layout (bytes, 256-aligned chunks) ---------
    char* base = (char*)d_ws;
    size_t off = 0;
    unsigned short* wp  = (unsigned short*)(base + off); off += (size_t)2 * L * HID * HID * sizeof(unsigned short);
    off = (off + 255) & ~(size_t)255;
    unsigned short* hbf = (unsigned short*)(base + off); off += (size_t)NH * sizeof(unsigned short);  // bf16 GEMM input
    off = (off + 255) & ~(size_t)255;
    float* bufB   = (float*)(base + off); off += (size_t)NH * sizeof(float);   // agg / GEMM output y
    float* bufC   = (float*)(base + off); off += (size_t)NH * sizeof(float);   // layer output ping
    float* bufD   = (float*)(base + off); off += (size_t)NH * sizeof(float);   // layer output pong
    float* subsum = (float*)(base + off); off += (size_t)S * 4 * HID * sizeof(float);
    float* subcnt = (float*)(base + off); off += ((size_t)S * sizeof(float) + 255) & ~(size_t)255;
    float* gsum   = (float*)(base + off); off += (size_t)G * 4 * HID * sizeof(float);
    float* gcnt   = (float*)(base + off); off += 256;
    float* ssum   = (float*)(base + off); off += 512;   // BN accumulators
    float* ssq    = (float*)(base + off); off += 512;
    float* mu     = (float*)(base + off); off += 512;
    float* rs     = (float*)(base + off); off += 512;

    const int BS = 256;
    const int* src = ei;
    const int* dst = ei + E;

    // zero accumulators (harness poisons ws; we own initialization)
    fillz_kernel<<<blocks_for((long long)S * 4 * HID, BS), BS, 0, stream>>>(subsum, (long long)S * 4 * HID);
    fillz_kernel<<<blocks_for(S, BS), BS, 0, stream>>>(subcnt, S);
    fillz_kernel<<<blocks_for((long long)G * 4 * HID, BS), BS, 0, stream>>>(gsum, (long long)G * 4 * HID);
    fillz_kernel<<<blocks_for(G, BS), BS, 0, stream>>>(gcnt, G);

    // pre-pack weights to bf16 B-fragments
    long long wtot = (long long)2 * L * HID * HID;
    pack_weights_kernel<<<blocks_for(wtot, BS), BS, 0, stream>>>(W1, W2, wp, L);

    // segment counts
    count_seg_kernel<<<blocks_for(N, BS), BS, 0, stream>>>(n2s, N, subcnt);
    count_seg_kernel<<<blocks_for(S, BS), BS, 0, stream>>>(s2g, S, gcnt);

    const float* Xcur = x;
    const long long Evec = (long long)E * 32;   // 4 floats per thread
    const long long Nvec = (long long)N * 32;
    const unsigned gemm_blocks = (unsigned)((N + 31) / 32);
    unsigned bn_blocks = blocks_for(NH, BS);
    if (bn_blocks > 4096) bn_blocks = 4096;

    for (int l = 0; l < L; ++l) {
        // neighbor-sum aggregation (f32)
        fillz_kernel<<<blocks_for(NH, BS), BS, 0, stream>>>(bufB, NH);
        scatter_agg_kernel<<<blocks_for(Evec, BS), BS, 0, stream>>>(Xcur, src, dst, bufB, Evec);
        combine_eps_bf16_kernel<<<blocks_for(NH, BS), BS, 0, stream>>>(Xcur, bufB, eps, l, hbf, NH);

        // MLP layer 1: GEMM (WMMA) + BN + ReLU -> bf16
        gemm_wmma_kernel<<<gemm_blocks, 256, 0, stream>>>(hbf, wp + (size_t)l * HID * HID,
                                                          b1 + (size_t)l * HID, bufB, N);
        fillz_kernel<<<1, BS, 0, stream>>>(ssum, 2 * HID);   // ssum+ssq contiguous
        bn_partial_kernel<<<bn_blocks, 256, 0, stream>>>(bufB, NH, ssum, ssq);
        bn_finalize_kernel<<<1, HID, 0, stream>>>(ssum, ssq, N, mu, rs);
        bn_apply_relu_bf16_kernel<<<blocks_for(NH, BS), BS, 0, stream>>>(bufB, mu, rs,
                g1 + (size_t)l * HID, be1 + (size_t)l * HID, hbf, NH);

        // MLP layer 2: GEMM (WMMA) + BN + ReLU -> f32 layer output
        gemm_wmma_kernel<<<gemm_blocks, 256, 0, stream>>>(hbf, wp + (size_t)(L + l) * HID * HID,
                                                          b2 + (size_t)l * HID, bufB, N);
        fillz_kernel<<<1, BS, 0, stream>>>(ssum, 2 * HID);
        bn_partial_kernel<<<bn_blocks, 256, 0, stream>>>(bufB, NH, ssum, ssq);
        bn_finalize_kernel<<<1, HID, 0, stream>>>(ssum, ssq, N, mu, rs);
        float* hout = (l & 1) ? bufD : bufC;
        bn_apply_relu_f32_kernel<<<blocks_for(NH, BS), BS, 0, stream>>>(bufB, mu, rs,
                g2 + (size_t)l * HID, be2 + (size_t)l * HID, hout, NH);

        // accumulate this layer's block of concat features into subgraph sums
        seg_accum_kernel<<<blocks_for(Nvec, BS), BS, 0, stream>>>(hout, n2s, subsum, l, Nvec);
        Xcur = hout;
    }

    // subgraph mean -> graph mean
    long long stot = (long long)S * 4 * HID;
    seg_div_kernel<<<blocks_for(stot, BS), BS, 0, stream>>>(subsum, subcnt, stot);
    graph_accum_kernel<<<blocks_for(stot, BS), BS, 0, stream>>>(subsum, s2g, gsum, stot);
    long long gtot = (long long)G * 4 * HID;
    seg_div_kernel<<<blocks_for(gtot, BS), BS, 0, stream>>>(gsum, gcnt, gtot);

    // classification head + log_softmax
    head_kernel<<<G, 128, 0, stream>>>(gsum, lin1_W, lin1_b, lin2_W, lin2_b, out);
}
